// SpatialTemporalMultiAgentModel_86552180949405
// MI455X (gfx1250) — matrix operand order repaired and checked by reading, not compiled, not run on hardware
//
#include <hip/hip_runtime.h>
#include <stdint.h>

// ---------------------------------------------------------------------------
// CDNA5 / gfx1250 implementation of SpatialTemporalMultiAgentModel forward.
// All large GEMMs run on v_wmma_f32_16x16x32_bf16 (bf16 operands, f32 acc).
// Model constants (from the reference): V=4096 E=128 H=512 NH=8 L=4 T=128
// N=64 B=4 WIN=16, M = B*T*N = 32768.
// ---------------------------------------------------------------------------

typedef __attribute__((ext_vector_type(16))) __bf16 v16bf;
typedef __attribute__((ext_vector_type(8)))  float  v8f;
typedef unsigned short u16b;   // raw bf16 bits

union Frag { uint4 u[2]; v16bf v; };
static_assert(sizeof(Frag) == 32, "frag size");

__device__ __forceinline__ u16b f2bf(float f) {
  union { float f; unsigned u; } c; c.f = f;
  unsigned u = c.u;
  unsigned r = 0x7fffu + ((u >> 16) & 1u);      // round-to-nearest-even
  return (u16b)((u + r) >> 16);
}
__device__ __forceinline__ float bf2f(u16b h) {
  union { unsigned u; float f; } c; c.u = ((unsigned)h) << 16;
  return c.f;
}
__device__ __forceinline__ float wave_sum(float v) {
#pragma unroll
  for (int m = 16; m >= 1; m >>= 1) v += __shfl_xor(v, m, 32);
  return v;
}
__device__ __forceinline__ float silu_f(float x) { return x / (1.0f + __expf(-x)); }

// ---------------------------------------------------------------------------
// Generic WMMA GEMM:  C[M,N] = A[M,K](bf16, row-major, lda=K)
//                            @ Wp(bf16 packed [K/16][N][16]) + bias[N]
// MODE 0: Cf = acc+bias            (f32 out)
// MODE 1: Cf += acc+bias           (f32 residual accumulate)
// MODE 2: Cb = bf16(acc+bias)      (bf16 out, e.g. q/k/v)
// MODE 3: Cb = bf16(silu(acc+bias))(bf16 out, FFN hidden)
// Block = 128 threads = 4 waves; each wave computes a 64x64 tile;
// block tile = 128(M) x 128(N). Requires M%128==0, N%128==0, K%32==0.
// MODE is a template parameter so each epilogue is branch-free straight-line
// code (the runtime-mode version compiled into a scalar-branch thicket).
// ---------------------------------------------------------------------------
template <int MODE>
__global__ __launch_bounds__(128)
void wmma_gemm_kernel(const u16b* __restrict__ A, const u16b* __restrict__ Wp,
                      const float* __restrict__ bias, float* __restrict__ Cf,
                      u16b* __restrict__ Cb, int M, int N, int K) {
  const int lane = threadIdx.x & 31;
  const int wave = threadIdx.x >> 5;
  const int wm = wave >> 1, wn = wave & 1;
  const int m0 = blockIdx.y * 128 + wm * 64;
  const int n0 = blockIdx.x * 128 + wn * 64;
  const int g   = lane >> 4;   // lane group (0: lanes 0-15, 1: lanes 16-31)
  const int l16 = lane & 15;

  v8f acc[4][4];
#pragma unroll
  for (int i = 0; i < 4; ++i)
#pragma unroll
    for (int j = 0; j < 4; ++j) {
#pragma unroll
      for (int r = 0; r < 8; ++r) acc[i][j][r] = 0.0f;
    }

  for (int kk = 0; kk < K; kk += 32) {
    Frag a[4];
#pragma unroll
    for (int mf = 0; mf < 4; ++mf) {
      const u16b* ap = A + (size_t)(m0 + mf * 16 + l16) * K + kk + 16 * g;
      a[mf].u[0] = *reinterpret_cast<const uint4*>(ap);        // K = kk+16g .. +7
      a[mf].u[1] = *reinterpret_cast<const uint4*>(ap + 8);    // K = kk+16g+8 .. +15
    }
    const int kb = (kk >> 4) + g;   // 16-row K block this lane group covers
#pragma unroll
    for (int nf = 0; nf < 4; ++nf) {
      Frag b;
      const u16b* bp = Wp + ((size_t)kb * N + (n0 + nf * 16 + l16)) * 16;
      b.u[0] = *reinterpret_cast<const uint4*>(bp);
      b.u[1] = *reinterpret_cast<const uint4*>(bp + 8);
#pragma unroll
      for (int mf = 0; mf < 4; ++mf)
        acc[mf][nf] = __builtin_amdgcn_wmma_f32_16x16x32_bf16(
            false, a[mf].v, false, b.v, (short)0, acc[mf][nf], false, false);
    }
  }

#pragma unroll
  for (int mf = 0; mf < 4; ++mf) {
#pragma unroll
    for (int nf = 0; nf < 4; ++nf) {
      const int col = n0 + nf * 16 + l16;
      const float bv = bias[col];
#pragma unroll
      for (int r = 0; r < 8; ++r) {
        const int row = m0 + mf * 16 + 8 * g + r;   // C layout: M = 8g + r
        const size_t idx = (size_t)row * N + col;
        float v = acc[mf][nf][r] + bv;
        if (MODE == 0)      Cf[idx] = v;
        else if (MODE == 1) Cf[idx] += v;
        else if (MODE == 2) Cb[idx] = f2bf(v);
        else                Cb[idx] = f2bf(silu_f(v));
      }
    }
  }
}

// ---------------------------------------------------------------------------
// Weight pack: f32 W[K,N] -> bf16 Wp[K/16][N][16]  (B-fragment native layout)
// ---------------------------------------------------------------------------
__global__ void pack_weight_kernel(const float* __restrict__ W,
                                   u16b* __restrict__ Wp, int K, int N) {
  int i = blockIdx.x * 256 + threadIdx.x;
  if (i >= K * N) return;
  int k = i / N, n = i - k * N;
  Wp[((size_t)(k >> 4) * N + n) * 16 + (k & 15)] = f2bf(W[i]);
}

// ---------------------------------------------------------------------------
// ce gather: ce[m, 0:128]=tok_emb[traj], [128:256]=tok_emb[cond0],
//            [256:384]=tok_emb[cond1]; bf16 out.
// ---------------------------------------------------------------------------
__global__ void gather_ce_kernel(const int* __restrict__ traj,
                                 const int* __restrict__ cond,
                                 const float* __restrict__ tok_emb,
                                 u16b* __restrict__ ce, int M) {
  int i = blockIdx.x * 256 + threadIdx.x;
  if (i >= M * 384) return;
  int m = i / 384, c = i - m * 384;
  int chunk = c >> 7, cc = c & 127;
  int tok = (chunk == 0) ? traj[m] : cond[m * 2 + chunk - 1];
  ce[i] = f2bf(tok_emb[tok * 128 + cc]);
}

// ---------------------------------------------------------------------------
// cond score: per token, score = softmax(h_act @ cw2 + cb2) over 3.
// One wave per token.
// ---------------------------------------------------------------------------
__global__ __launch_bounds__(128)
void cond_score_kernel(const float* __restrict__ Hact,
                       const float* __restrict__ cw2,
                       const float* __restrict__ cb2,
                       float* __restrict__ score, int M) {
  int wave = threadIdx.x >> 5, lane = threadIdx.x & 31;
  int row = blockIdx.x * 4 + wave;
  if (row >= M) return;
  const float* hr = Hact + (size_t)row * 512;
  float p0 = 0.f, p1 = 0.f, p2 = 0.f;
#pragma unroll 4
  for (int c = lane; c < 512; c += 32) {
    float h = hr[c];
    p0 += h * cw2[c * 3 + 0];
    p1 += h * cw2[c * 3 + 1];
    p2 += h * cw2[c * 3 + 2];
  }
  p0 = wave_sum(p0) + cb2[0];
  p1 = wave_sum(p1) + cb2[1];
  p2 = wave_sum(p2) + cb2[2];
  float mx = fmaxf(p0, fmaxf(p1, p2));
  float e0 = __expf(p0 - mx), e1 = __expf(p1 - mx), e2 = __expf(p2 - mx);
  float inv = 1.f / (e0 + e1 + e2);
  if (lane == 0) {
    score[row * 3 + 0] = e0 * inv;
    score[row * 3 + 1] = e1 * inv;
    score[row * 3 + 2] = e2 * inv;
  }
}

// ---------------------------------------------------------------------------
// x0 = tok_emb[traj]*(1+s0) + pos_emb[t] + s1*tok_emb[cond0] + s2*tok_emb[cond1]
// (cond_emb folded in); bf16 out (feeds the input-projection GEMM).
// ---------------------------------------------------------------------------
__global__ void build_x0_kernel(const int* __restrict__ traj,
                                const int* __restrict__ cond,
                                const float* __restrict__ tok_emb,
                                const float* __restrict__ pos_emb,
                                const float* __restrict__ score,
                                u16b* __restrict__ x0, int M) {
  int i = blockIdx.x * 256 + threadIdx.x;
  if (i >= M * 128) return;
  int m = i >> 7, c = i & 127;
  int t = (m / 64) % 128;                 // m = (b*T + t)*N + n
  float s0 = score[m * 3 + 0], s1 = score[m * 3 + 1], s2 = score[m * 3 + 2];
  float v = tok_emb[traj[m] * 128 + c] * (1.f + s0)
          + pos_emb[t * 128 + c]
          + s1 * tok_emb[cond[m * 2 + 0] * 128 + c]
          + s2 * tok_emb[cond[m * 2 + 1] * 128 + c];
  x0[i] = f2bf(v);
}

// ---------------------------------------------------------------------------
// LayerNorm over H=512 per row. MODE 0: f32 out = silu(ln(x)) (in-place ok)
//                               MODE 1: bf16 out = ln(x)
// 256 threads = 8 waves, one row per wave.
// ---------------------------------------------------------------------------
template <int MODE>
__global__ __launch_bounds__(256)
void ln_kernel(const float* __restrict__ X, const float* __restrict__ gain,
               const float* __restrict__ beta, float* __restrict__ outF,
               u16b* __restrict__ outB, int M) {
  const int wave = threadIdx.x >> 5, lane = threadIdx.x & 31;
  const int row = blockIdx.x * 8 + wave;
  if (row >= M) return;
  const float* xr = X + (size_t)row * 512;
  float s = 0.f, s2 = 0.f;
#pragma unroll
  for (int k = 0; k < 16; ++k) {
    float v = xr[lane + 32 * k];
    s += v; s2 += v * v;
  }
  s = wave_sum(s); s2 = wave_sum(s2);
  float mean = s * (1.f / 512.f);
  float var  = s2 * (1.f / 512.f) - mean * mean;
  float rinv = rsqrtf(var + 1e-5f);
#pragma unroll
  for (int k = 0; k < 16; ++k) {
    int c = lane + 32 * k;
    float y = (xr[c] - mean) * rinv * gain[c] + beta[c];
    if (MODE == 0) outF[(size_t)row * 512 + c] = silu_f(y);
    else           outB[(size_t)row * 512 + c] = f2bf(y);
  }
}

// ---------------------------------------------------------------------------
// Temporal attention core (per (b,n) sequence of T=128, per head, d=64),
// ALiBi + sliding-window bias, online softmax. Q/K/V/O bf16 [M,512].
// Block = 128 threads (one query row per thread). LDS: K,V transposed [d][j]
// (conflict-free writes, broadcast reads).
// ---------------------------------------------------------------------------
__global__ __launch_bounds__(128)
void attn_temporal_kernel(const u16b* __restrict__ Q, const u16b* __restrict__ Kb,
                          const u16b* __restrict__ Vb, u16b* __restrict__ O) {
  __shared__ float skT[64][128];
  __shared__ float svT[64][128];
  const int p = blockIdx.x;           // b*N + n  (0..255)
  const int h = blockIdx.y;
  const int b = p >> 6;
  const int n = p & 63;
  const int i = threadIdx.x;          // query timestep t = i
  const size_t off = ((size_t)((b * 128 + i) * 64 + n)) * 512 + h * 64;
  float qreg[64];
#pragma unroll
  for (int d = 0; d < 64; ++d) {
    qreg[d]  = bf2f(Q[off + d]);
    skT[d][i] = bf2f(Kb[off + d]);
    svT[d][i] = bf2f(Vb[off + d]);
  }
  __syncthreads();
  const float slope = exp2f(-(float)(h + 1));   // 2^(-8/NH*(h+1)), NH=8
  float m_run = -__builtin_inff(), l = 0.f;
  float o[64];
#pragma unroll
  for (int d = 0; d < 64; ++d) o[d] = 0.f;
  for (int j = 0; j < 128; ++j) {
    float s = 0.f;
#pragma unroll
    for (int d = 0; d < 64; ++d) s += qreg[d] * skT[d][j];
    const int dg = j - i;
    const float bias = (dg <= 0 && dg > -16) ? slope * (float)dg : -1.0e9f;
    s = s * 0.125f + bias;                      // 1/sqrt(64)
    const float mnew  = fmaxf(m_run, s);
    const float scale = __expf(m_run - mnew);
    const float e     = __expf(s - mnew);
    l = l * scale + e;
#pragma unroll
    for (int d = 0; d < 64; ++d) o[d] = o[d] * scale + e * svT[d][j];
    m_run = mnew;
  }
  const float inv = 1.f / l;
#pragma unroll
  for (int d = 0; d < 64; ++d) O[off + d] = f2bf(o[d] * inv);
}

// ---------------------------------------------------------------------------
// Spatial (cross-agent) attention core (per (b,t), S=N=64, per head, d=64),
// reagent-mask bias. Block = 64 threads.
// ---------------------------------------------------------------------------
__global__ __launch_bounds__(64)
void attn_spatial_kernel(const u16b* __restrict__ Q, const u16b* __restrict__ Kb,
                         const u16b* __restrict__ Vb, const int* __restrict__ mask,
                         u16b* __restrict__ O) {
  __shared__ float skT[64][64];
  __shared__ float svT[64][64];
  __shared__ int mv[64];
  const int bt = blockIdx.x;          // b*T + t (0..511)
  const int h  = blockIdx.y;
  const int i  = threadIdx.x;         // agent index
  const size_t off = ((size_t)(bt * 64 + i)) * 512 + h * 64;
  float qreg[64];
#pragma unroll
  for (int d = 0; d < 64; ++d) {
    qreg[d]  = bf2f(Q[off + d]);
    skT[d][i] = bf2f(Kb[off + d]);
    svT[d][i] = bf2f(Vb[off + d]);
  }
  mv[i] = mask[bt * 64 + i];
  __syncthreads();
  const int mi = mv[i];
  float m_run = -__builtin_inff(), l = 0.f;
  float o[64];
#pragma unroll
  for (int d = 0; d < 64; ++d) o[d] = 0.f;
  for (int j = 0; j < 64; ++j) {
    float s = 0.f;
#pragma unroll
    for (int d = 0; d < 64; ++d) s += qreg[d] * skT[d][j];
    const float bias = (mi && mv[j]) ? 0.f : -1.0e9f;
    s = s * 0.125f + bias;
    const float mnew  = fmaxf(m_run, s);
    const float scale = __expf(m_run - mnew);
    const float e     = __expf(s - mnew);
    l = l * scale + e;
#pragma unroll
    for (int d = 0; d < 64; ++d) o[d] = o[d] * scale + e * svT[d][j];
    m_run = mnew;
  }
  const float inv = 1.f / l;
#pragma unroll
  for (int d = 0; d < 64; ++d) O[off + d] = f2bf(o[d] * inv);
}

// ---------------------------------------------------------------------------
// Host orchestration.
// d_in layout: [0]=traj (B,T,N) int, [1]=cond (B,T,N,2) int,
// [2]=reagent_mask (B,T,N) int, then params leaves in JAX pytree order
// (dict keys sorted lexicographically at every level):
//   3:bf 4:cb1 5:cb2 6:cbn1 7:cg1 8:cw1 9:cw2 10:gf 11:hb 12:hw
//   13:ib 14:ibn 15:ig 16:iw
//   17 + l*26 + k for layer l, k in:
//     0:b1 1:b2 2:b3 3:fb1 4:fb2 5:fw1 6:fw2 7:g1 8:g2 9:g3
//     10:s.bk 11:s.bo 12:s.bq 13:s.bv 14:s.wk 15:s.wo 16:s.wq 17:s.wv
//     18:t.bk 19:t.bo 20:t.bq 21:t.bv 22:t.wk 23:t.wo 24:t.wq 25:t.wv
//   121:pos_emb 122:tok_emb
// ---------------------------------------------------------------------------
extern "C" void kernel_launch(void* const* d_in, const int* in_sizes, int n_in,
                              void* d_out, int out_size, void* d_ws, size_t ws_size,
                              hipStream_t stream) {
  (void)in_sizes; (void)n_in; (void)out_size; (void)ws_size;
  const int M = 32768;   // B*T*N

  const int* traj = (const int*)d_in[0];
  const int* cond = (const int*)d_in[1];
  const int* rmask = (const int*)d_in[2];
  auto P = [&](int i) { return (const float*)d_in[i]; };
  auto LP = [&](int l, int k) { return (const float*)d_in[17 + l * 26 + k]; };
  const float* tok_emb = P(122);
  const float* pos_emb = P(121);

  // ---- workspace carve (≈360 MB) ----
  char* base = (char*)d_ws;
  size_t off = 0;
  auto alloc = [&](size_t bytes) -> char* {
    char* p = base + off;
    off += (bytes + 255) & ~(size_t)255;
    return p;
  };
  u16b* cw1p = (u16b*)alloc((size_t)384 * 512 * 2);
  u16b* iwp  = (u16b*)alloc((size_t)128 * 512 * 2);
  u16b* hwp  = (u16b*)alloc((size_t)512 * 4096 * 2);
  u16b* lw[4][10];   // twq twk twv two swq swk swv swo fw1 fw2
  for (int l = 0; l < 4; ++l) {
    for (int k = 0; k < 8; ++k) lw[l][k] = (u16b*)alloc((size_t)512 * 512 * 2);
    lw[l][8] = (u16b*)alloc((size_t)512 * 2048 * 2);
    lw[l][9] = (u16b*)alloc((size_t)2048 * 512 * 2);
  }
  float* XF   = (float*)alloc((size_t)M * 512 * 4);   // f32 residual stream
  u16b* A1    = (u16b*)alloc((size_t)M * 512 * 2);    // bf16 staging (ce/x0/ln/attn-out)
  u16b* FFB   = (u16b*)alloc((size_t)M * 2048 * 2);   // bf16 FFN hidden
  u16b* QB    = (u16b*)alloc((size_t)M * 512 * 2);
  u16b* KB    = (u16b*)alloc((size_t)M * 512 * 2);
  u16b* VB    = (u16b*)alloc((size_t)M * 512 * 2);
  float* SCOREB = (float*)alloc((size_t)M * 3 * 4);

  auto pack = [&](const float* src, u16b* dst, int K, int N) {
    int total = K * N;
    pack_weight_kernel<<<(total + 255) / 256, 256, 0, stream>>>(src, dst, K, N);
  };
  auto gemm = [&](const u16b* A, const u16b* Wp, const float* bias, float* Cf,
                  u16b* Cb, int Mm, int Nn, int Kk, int mode) {
    dim3 grid(Nn / 128, Mm / 128);
    switch (mode) {
      case 0: wmma_gemm_kernel<0><<<grid, 128, 0, stream>>>(A, Wp, bias, Cf, Cb, Mm, Nn, Kk); break;
      case 1: wmma_gemm_kernel<1><<<grid, 128, 0, stream>>>(A, Wp, bias, Cf, Cb, Mm, Nn, Kk); break;
      case 2: wmma_gemm_kernel<2><<<grid, 128, 0, stream>>>(A, Wp, bias, Cf, Cb, Mm, Nn, Kk); break;
      default: wmma_gemm_kernel<3><<<grid, 128, 0, stream>>>(A, Wp, bias, Cf, Cb, Mm, Nn, Kk); break;
    }
  };
  auto ln_silu_f32 = [&](const float* X, const float* g, const float* b, float* oF) {
    ln_kernel<0><<<M / 8, 256, 0, stream>>>(X, g, b, oF, nullptr, M);
  };
  auto ln_bf16 = [&](const float* X, const float* g, const float* b, u16b* oB) {
    ln_kernel<1><<<M / 8, 256, 0, stream>>>(X, g, b, nullptr, oB, M);
  };

  // ---- pack all weights to bf16 (B-frag layout) ----
  pack(P(8),  cw1p, 384, 512);     // cw1
  pack(P(16), iwp,  128, 512);     // iw
  pack(P(12), hwp,  512, 4096);    // hw
  static const int widx[8] = {24, 22, 25, 23, 16, 14, 17, 15};
  for (int l = 0; l < 4; ++l) {
    for (int k = 0; k < 8; ++k) pack(LP(l, widx[k]), lw[l][k], 512, 512);
    pack(LP(l, 5), lw[l][8], 512, 2048);   // fw1
    pack(LP(l, 6), lw[l][9], 2048, 512);   // fw2
  }

  // ---- conditioning MLP ----
  gather_ce_kernel<<<(M * 384 + 255) / 256, 256, 0, stream>>>(traj, cond, tok_emb, A1, M);
  gemm(A1, cw1p, P(4), XF, nullptr, M, 512, 384, 0);          // h = ce@cw1 + cb1
  ln_silu_f32(XF, P(7), P(6), XF);                            // h = silu(ln(h; cg1, cbn1))
  cond_score_kernel<<<M / 4, 128, 0, stream>>>(XF, P(9), P(5), SCOREB, M);
  build_x0_kernel<<<(M * 128 + 255) / 256, 256, 0, stream>>>(traj, cond, tok_emb,
                                                             pos_emb, SCOREB, A1, M);
  gemm(A1, iwp, P(13), XF, nullptr, M, 512, 128, 0);          // x = x0@iw + ib
  ln_silu_f32(XF, P(15), P(14), XF);                          // x = silu(ln(x; ig, ibn))

  // ---- transformer layers ----
  for (int l = 0; l < 4; ++l) {
    // temporal attention
    ln_bf16(XF, LP(l, 7), LP(l, 0), A1);                      // ln(x; g1,b1) -> bf16
    gemm(A1, lw[l][0], LP(l, 20), nullptr, QB, M, 512, 512, 2);  // q = . @wq + bq
    gemm(A1, lw[l][1], LP(l, 18), nullptr, KB, M, 512, 512, 2);  // k = . @wk + bk
    gemm(A1, lw[l][2], LP(l, 21), nullptr, VB, M, 512, 512, 2);  // v = . @wv + bv
    attn_temporal_kernel<<<dim3(256, 8), 128, 0, stream>>>(QB, KB, VB, A1);
    gemm(A1, lw[l][3], LP(l, 19), XF, nullptr, M, 512, 512, 1);  // x += o@wo + bo

    // spatial attention
    ln_bf16(XF, LP(l, 8), LP(l, 1), A1);                      // ln(x; g2,b2)
    gemm(A1, lw[l][4], LP(l, 12), nullptr, QB, M, 512, 512, 2);
    gemm(A1, lw[l][5], LP(l, 10), nullptr, KB, M, 512, 512, 2);
    gemm(A1, lw[l][6], LP(l, 13), nullptr, VB, M, 512, 512, 2);
    attn_spatial_kernel<<<dim3(512, 8), 64, 0, stream>>>(QB, KB, VB, rmask, A1);
    gemm(A1, lw[l][7], LP(l, 11), XF, nullptr, M, 512, 512, 1);

    // feed-forward
    ln_bf16(XF, LP(l, 9), LP(l, 2), A1);                      // ln(x; g3,b3)
    gemm(A1, lw[l][8], LP(l, 3), nullptr, FFB, M, 2048, 512, 3);  // silu(.@fw1+fb1)
    gemm(FFB, lw[l][9], LP(l, 4), XF, nullptr, M, 512, 2048, 1);  // x += .@fw2+fb2
  }

  // ---- head ----
  ln_bf16(XF, P(10), P(3), A1);                               // ln(x; gf, bf)
  gemm(A1, hwp, P(11), (float*)d_out, nullptr, M, 4096, 512, 0);  // logits
}
